// CausalMultiHeadSelfAttention_18605798326746
// MI455X (gfx1250) — matrix operand order repaired and checked by reading, not compiled, not run on hardware
//
#include <hip/hip_runtime.h>

typedef __bf16 bf16;
typedef __attribute__((ext_vector_type(16))) __bf16 v16bf;
typedef __attribute__((ext_vector_type(8)))  __bf16 v8bf;
typedef __attribute__((ext_vector_type(8)))  float  v8f;
typedef __attribute__((ext_vector_type(4)))  unsigned int u32x4;
typedef __attribute__((ext_vector_type(8)))  int i32x8;
typedef __attribute__((ext_vector_type(4)))  int i32x4;

union Frag16 { v16bf v; v8bf h[2]; };
union CFrag  { v8f  v; float f[8]; };

#define D_MODEL 1024
#define SEQ     2048
#define BATCH   4
#define NH      16
#define DH      64

// ---------------------------------------------------------------------------
// Fragment loaders (CDNA5 wave32 WMMA 16x16x32 bf16 layouts, CDNA5 ISA)
// A (16x32, MxK): lane L: M=L&15; lanes<16: v0..3=K0..7, v4..7=K16..23;
//                 lanes>=16: v0..3=K8..15, v4..7=K24..31.
// B (32x16, KxN): lane L: N=L&15; lanes<16: v0..3=K0..7, v4..7=K16..23;
//                 lanes>=16: v0..3=K8..15, v4..7=K24..31.
// C (16x16 f32): VGPR r: lanes<16 -> M=r, lanes>=16 -> M=r+8; N=L&15.
// ---------------------------------------------------------------------------
__device__ __forceinline__ v16bf load_frag_a(const bf16* base, int ld) {
  const int lane = threadIdx.x & 31;
  const int m  = lane & 15;
  const int kh = (lane >> 4) * 8;
  const bf16* row = base + m * ld;
  Frag16 f;
  f.h[0] = *(const v8bf*)(row + kh);
  f.h[1] = *(const v8bf*)(row + 16 + kh);
  return f.v;
}

__device__ __forceinline__ v16bf load_frag_b(const bf16* base, int ld) {
  const int lane = threadIdx.x & 31;
  const int n  = lane & 15;
  const int kh = (lane >> 4) * 8;
  const bf16* row = base + n * ld;
  Frag16 f;
  f.h[0] = *(const v8bf*)(row + kh);
  f.h[1] = *(const v8bf*)(row + 16 + kh);
  return f.v;
}

// ---------------------------------------------------------------------------
// TDM: issue one tensor_load_to_lds for a 2-D [rows x 32] bf16 tile.
// D# per CDNA5 ISA.
// pad: interval=3 (16 DWORDs = one 32-elem bf16 row), amount=3 (4 DWORDs =
//      8 bf16 elems) -> padded LDS leading dim of 40 elements.
// Toolchain here is the 6-arg clang-23 builtin:
//   (u32x4 g0, i32x8 g1, i32x4 g2, i32x4 g3, i32x8 extra, i32 cpol)
// ---------------------------------------------------------------------------
__device__ __forceinline__ void tdm_load_tile_2d(
    const bf16* gptr, unsigned lds_byte_off,
    unsigned tensor_d0_elems, unsigned rows,
    unsigned row_stride_elems)
{
  unsigned long long ga = (unsigned long long)(uintptr_t)gptr;
  u32x4 g0;
  g0[0] = 1u;                                           // count=1 (valid, user)
  g0[1] = lds_byte_off;                                 // lds_addr
  g0[2] = (unsigned)(ga & 0xffffffffu);                 // global_addr[31:0]
  g0[3] = (unsigned)((ga >> 32) & 0x01ffffffu) | (2u << 30); // addr[56:32]|type=2

  i32x8 g1;
  unsigned w0 = (1u << 16)                              // data_size=1 (2 bytes)
              | (1u << 20)                              // pad_enable
              | (3u << 22)                              // pad_interval: 16 DW
              | (3u << 25);                             // pad_amount: 4 DW
  g1[0] = (int)w0;
  g1[1] = (int)((tensor_d0_elems & 0xffffu) << 16);     // [63:48]=tensor_dim0 lo
  g1[2] = (int)((tensor_d0_elems >> 16) | ((rows & 0xffffu) << 16)); // dim0 hi | dim1 lo
  g1[3] = (int)((rows >> 16) | (32u << 16));            // dim1 hi | tile_dim0=32
  g1[4] = (int)(rows & 0xffffu);                        // tile_dim1=rows, tile_dim2=0
  g1[5] = (int)row_stride_elems;                        // tensor_dim0_stride lo32
  g1[6] = 0;                                            // stride hi16 | dim1_stride lo
  g1[7] = 0;
  i32x4 z4 = {0, 0, 0, 0};                              // 2-D tensor: groups 2/3 unused
  i32x8 z8 = {0, 0, 0, 0, 0, 0, 0, 0};
  __builtin_amdgcn_tensor_load_to_lds(g0, g1, z4, z4, z8, 0);
}

// ---------------------------------------------------------------------------
// fp32 -> bf16 convert
// ---------------------------------------------------------------------------
__global__ void f32_to_bf16_kernel(const float* __restrict__ in,
                                   bf16* __restrict__ out, int n) {
  int i = blockIdx.x * blockDim.x + threadIdx.x;
  if (i < n) out[i] = (bf16)in[i];
}

// ---------------------------------------------------------------------------
// Tiled GEMM:  C[M,N] = A[M,K] * W[N,K]^T (bf16, row-major), TDM-staged tiles.
// Macro-tile 128x64, BK=32, 8 waves; wave w computes rows [w*16, w*16+16) x 64.
// mode 0: fp32 C -> outF ; mode 1: scatter bf16 to Qb/Kb [B,H,S,dh], Vt [B,H,dh,S]
// ---------------------------------------------------------------------------
__global__ __launch_bounds__(256) void gemm_bf16_nt(
    const bf16* __restrict__ A, const bf16* __restrict__ W,
    int M, int N, int K, int mode,
    float* __restrict__ outF,
    bf16* __restrict__ Qb, bf16* __restrict__ Kb, bf16* __restrict__ Vt)
{
  __shared__ __align__(16) bf16 As[128 * 40];
  __shared__ __align__(16) bf16 Ws[64 * 40];

  const int tid  = threadIdx.x;
  const int lane = tid & 31;
  const int wave = tid >> 5;
  const int m0 = blockIdx.x * 128;
  const int n0 = blockIdx.y * 64;
  const int msub = wave * 16;

  const unsigned ldsA = (unsigned)(uintptr_t)(void*)&As[0];  // low 32 bits = LDS offset
  const unsigned ldsW = (unsigned)(uintptr_t)(void*)&Ws[0];

  CFrag c[4];
#pragma unroll
  for (int t = 0; t < 4; ++t)
#pragma unroll
    for (int i = 0; i < 8; ++i) c[t].f[i] = 0.f;

  for (int k0 = 0; k0 < K; k0 += 32) {
    if (wave == 0) {
      tdm_load_tile_2d(A + (size_t)m0 * K + k0, ldsA, (unsigned)K, 128, (unsigned)K);
      tdm_load_tile_2d(W + (size_t)n0 * K + k0, ldsW, (unsigned)K, 64,  (unsigned)K);
      __builtin_amdgcn_s_wait_tensorcnt(0);
    }
    __syncthreads();
    v16bf a = load_frag_a(As + msub * 40, 40);
#pragma unroll
    for (int t = 0; t < 4; ++t) {
      v16bf b = load_frag_b(Ws + t * 16 * 40, 40);
      c[t].v = __builtin_amdgcn_wmma_f32_16x16x32_bf16(false, a, false, b,
                                                       (short)0, c[t].v, false, false);
    }
    __syncthreads();
  }

  const int hi8 = (lane >> 4) ? 8 : 0;
  const int nl  = lane & 15;
#pragma unroll
  for (int t = 0; t < 4; ++t) {
#pragma unroll
    for (int r = 0; r < 8; ++r) {
      const int row = m0 + msub + r + hi8;
      const int col = n0 + t * 16 + nl;
      if (mode == 0) {
        outF[row * N + col] = c[t].f[r];
      } else {
        const int b = row >> 11, s = row & 2047;
        const int which = col >> 10;
        const int e = col & 1023;
        const int h = e >> 6;
        const int d = e & 63;
        const int bh = b * NH + h;
        const float val = c[t].f[r];
        if (which == 0)      Qb[(bh * SEQ + s) * DH + d] = (bf16)val;
        else if (which == 1) Kb[(bh * SEQ + s) * DH + d] = (bf16)val;
        else                 Vt[(bh * DH + d) * SEQ + s] = (bf16)val;
      }
    }
  }
}

// ---------------------------------------------------------------------------
// RoPE in-place on Q,K [B,H,S,dh]; positions = s (arange), interleaved pairs
// ---------------------------------------------------------------------------
__global__ void rope_kernel(bf16* __restrict__ Qb, bf16* __restrict__ Kb) {
  int idx = blockIdx.x * blockDim.x + threadIdx.x;
  if (idx >= BATCH * NH * SEQ * (DH / 2)) return;
  const int i  = idx & 31;
  const int s  = (idx >> 5) & 2047;
  const int bh = idx >> 16;
  const float inv = __expf(-(float)i * (9.210340371976184f / 32.0f)); // theta^(-2i/64)
  const float ang = (float)s * inv;
  float sn, cs;
  __sincosf(ang, &sn, &cs);
  const int off = (bh * SEQ + s) * DH + 2 * i;
  {
    float x1 = (float)Qb[off], x2 = (float)Qb[off + 1];
    Qb[off]     = (bf16)(x1 * cs - x2 * sn);
    Qb[off + 1] = (bf16)(x1 * sn + x2 * cs);
  }
  {
    float x1 = (float)Kb[off], x2 = (float)Kb[off + 1];
    Kb[off]     = (bf16)(x1 * cs - x2 * sn);
    Kb[off + 1] = (bf16)(x1 * sn + x2 * cs);
  }
}

// ---------------------------------------------------------------------------
// Flash attention. Block = 4 waves; wave w owns query rows qbase..qbase+15 of
// one (b,h). Key tiles of 32; scores stay in registers; P goes through a
// private per-wave LDS tile to re-layout C-frag -> A-frag for the PV WMMAs.
// ---------------------------------------------------------------------------
__global__ __launch_bounds__(128) void flash_attn_kernel(
    const bf16* __restrict__ Qb, const bf16* __restrict__ Kb,
    const bf16* __restrict__ Vt, bf16* __restrict__ Ob)
{
  __shared__ __align__(16) bf16 Ptile[4 * 16 * 40];

  const int lane = threadIdx.x & 31;
  const int wave = threadIdx.x >> 5;
  const int bh    = blockIdx.y;
  const int qbase = blockIdx.x * 64 + wave * 16;

  const bf16* Q  = Qb + bh * SEQ * DH;
  const bf16* Kp = Kb + bh * SEQ * DH;
  const bf16* Vp = Vt + bh * DH * SEQ;
  bf16* pw = Ptile + wave * 16 * 40;

  const int hi8 = (lane >> 4) ? 8 : 0;
  const int nl  = lane & 15;

  v16bf qa[2];
#pragma unroll
  for (int ks = 0; ks < 2; ++ks)
    qa[ks] = load_frag_a(Q + qbase * DH + ks * 32, DH);

  float mrow[8], lrow[8];
  CFrag o[4];
#pragma unroll
  for (int r = 0; r < 8; ++r) { mrow[r] = -__builtin_inff(); lrow[r] = 0.f; }
#pragma unroll
  for (int t = 0; t < 4; ++t)
#pragma unroll
    for (int r = 0; r < 8; ++r) o[t].f[r] = 0.f;

  for (int j = 0; j <= qbase + 15; j += 32) {    // causal, wave-uniform bound
    CFrag sc[2];
#pragma unroll
    for (int sub = 0; sub < 2; ++sub) {
#pragma unroll
      for (int i2 = 0; i2 < 8; ++i2) sc[sub].f[i2] = 0.f;
#pragma unroll
      for (int ks = 0; ks < 2; ++ks) {
        v16bf bk = load_frag_b(Kp + (j + sub * 16) * DH + ks * 32, DH);
        sc[sub].v = __builtin_amdgcn_wmma_f32_16x16x32_bf16(
            false, qa[ks], false, bk, (short)0, sc[sub].v, false, false);
      }
    }
#pragma unroll
    for (int sub = 0; sub < 2; ++sub) {
      const int col = j + sub * 16 + nl;
#pragma unroll
      for (int r = 0; r < 8; ++r) {
        const int rowq = qbase + r + hi8;
        const float sv = sc[sub].f[r] * 0.125f;   // 1/sqrt(64)
        sc[sub].f[r] = (col <= rowq) ? sv : -__builtin_inff();
      }
    }
#pragma unroll
    for (int r = 0; r < 8; ++r) {
      float mx = fmaxf(sc[0].f[r], sc[1].f[r]);
#pragma unroll
      for (int offm = 1; offm < 16; offm <<= 1)
        mx = fmaxf(mx, __shfl_xor(mx, offm, 16));
      const float mnew  = fmaxf(mrow[r], mx);
      const float alpha = __expf(mrow[r] - mnew);
      mrow[r] = mnew;
      const float p0 = __expf(sc[0].f[r] - mnew);
      const float p1 = __expf(sc[1].f[r] - mnew);
      float ps = p0 + p1;
#pragma unroll
      for (int offm = 1; offm < 16; offm <<= 1)
        ps += __shfl_xor(ps, offm, 16);
      lrow[r] = lrow[r] * alpha + ps;
      sc[0].f[r] = p0;
      sc[1].f[r] = p1;
#pragma unroll
      for (int t = 0; t < 4; ++t) o[t].f[r] *= alpha;
    }
#pragma unroll
    for (int sub = 0; sub < 2; ++sub)
#pragma unroll
      for (int r = 0; r < 8; ++r)
        pw[(r + hi8) * 40 + sub * 16 + nl] = (bf16)sc[sub].f[r];
    asm volatile("s_wait_dscnt 0x0" ::: "memory");  // wave-local LDS RAW
    v16bf pa = load_frag_a(pw, 40);
#pragma unroll
    for (int t = 0; t < 4; ++t) {
      v16bf vb = load_frag_b(Vp + (t * 16) * SEQ + j, SEQ);
      o[t].v = __builtin_amdgcn_wmma_f32_16x16x32_bf16(
          false, pa, false, vb, (short)0, o[t].v, false, false);
    }
  }

  const int b = bh >> 4, h = bh & 15;
#pragma unroll
  for (int t = 0; t < 4; ++t)
#pragma unroll
    for (int r = 0; r < 8; ++r) {
      const int s = qbase + r + hi8;
      const int d = t * 16 + nl;
      Ob[(b * SEQ + s) * D_MODEL + h * DH + d] = (bf16)(o[t].f[r] / lrow[r]);
    }
}

// ---------------------------------------------------------------------------
extern "C" void kernel_launch(void* const* d_in, const int* in_sizes, int n_in,
                              void* d_out, int out_size, void* d_ws, size_t ws_size,
                              hipStream_t stream) {
  const float* x  = (const float*)d_in[0];
  const float* wq = (const float*)d_in[1];
  const float* wk = (const float*)d_in[2];
  const float* wv = (const float*)d_in[3];
  const float* wo = (const float*)d_in[4];
  float* out = (float*)d_out;

  const size_t NTOK = (size_t)BATCH * SEQ;        // 8192 tokens
  char* ws = (char*)d_ws;
  size_t off = 0;
  auto alloc = [&](size_t bytes) -> void* {
    void* p = ws + off;
    off = (off + bytes + 255) & ~(size_t)255;
    return p;
  };
  bf16* xb    = (bf16*)alloc(NTOK * D_MODEL * 2);
  bf16* wqkvb = (bf16*)alloc(3ull * D_MODEL * D_MODEL * 2);
  bf16* wob   = (bf16*)alloc((size_t)D_MODEL * D_MODEL * 2);
  bf16* Qb    = (bf16*)alloc(NTOK * D_MODEL * 2);
  bf16* Kb    = (bf16*)alloc(NTOK * D_MODEL * 2);
  bf16* Vt    = (bf16*)alloc(NTOK * D_MODEL * 2);
  bf16* Ob    = (bf16*)alloc(NTOK * D_MODEL * 2);

  const int nElemX = (int)(NTOK * D_MODEL);
  const int nElemW = D_MODEL * D_MODEL;
  f32_to_bf16_kernel<<<(nElemX + 255) / 256, 256, 0, stream>>>(x, xb, nElemX);
  f32_to_bf16_kernel<<<(nElemW + 255) / 256, 256, 0, stream>>>(wq, wqkvb + 0 * nElemW, nElemW);
  f32_to_bf16_kernel<<<(nElemW + 255) / 256, 256, 0, stream>>>(wk, wqkvb + 1 * nElemW, nElemW);
  f32_to_bf16_kernel<<<(nElemW + 255) / 256, 256, 0, stream>>>(wv, wqkvb + 2 * nElemW, nElemW);
  f32_to_bf16_kernel<<<(nElemW + 255) / 256, 256, 0, stream>>>(wo, wob, nElemW);

  // fused QKV projection: [8192,1024] x [3072,1024]^T, scattered to Q/K/Vt
  gemm_bf16_nt<<<dim3(8192 / 128, 3072 / 64), 256, 0, stream>>>(
      xb, wqkvb, 8192, 3072, 1024, 1, nullptr, Qb, Kb, Vt);

  const int nPairs = BATCH * NH * SEQ * (DH / 2);
  rope_kernel<<<(nPairs + 255) / 256, 256, 0, stream>>>(Qb, Kb);

  flash_attn_kernel<<<dim3(SEQ / 64, BATCH * NH), 128, 0, stream>>>(Qb, Kb, Vt, Ob);

  // output projection -> fp32 d_out
  gemm_bf16_nt<<<dim3(8192 / 128, 1024 / 64), 256, 0, stream>>>(
      Ob, wob, 8192, 1024, 1024, 0, out, nullptr, nullptr, nullptr);
}
